// SelfAttention_77549929497067
// MI455X (gfx1250) — matrix operand order repaired and checked by reading, not compiled, not run on hardware
//
#include <hip/hip_runtime.h>
#include <hip/hip_bf16.h>
#include <stdint.h>

typedef __bf16 bf16_t;
typedef __attribute__((ext_vector_type(16))) __bf16 v16bf;
typedef __attribute__((ext_vector_type(8)))  float  v8f;

// ---------------------------------------------------------------------------
// Fragment loaders matching CDNA5 WMMA 16-bit VGPR layouts (CDNA5 ISA)
// A 16x32 (MxK): lane L: m=L%16, g=L/16; VGPR v holds K=(v<4?0:16)+g*8+2*(v%4),+1
// B 32x16 (KxN): lane L: n=L%16, kh=L/16; VGPR v holds K=(v<4?0:16)+kh*8+2*(v%4),+1
// Both read 2 consecutive bf16 (one dword) per VGPR from row-major storage,
// which the compiler coalesces into b64/b128 vector loads.
// ---------------------------------------------------------------------------
union FragU { v16bf v; uint32_t u[8]; };

__device__ __forceinline__ v16bf load_afrag(const bf16_t* __restrict__ X, int ld,
                                            int m0, int k0) {
  int lane = threadIdx.x & 31;
  int m = lane & 15, g = lane >> 4;
  const bf16_t* row = X + (size_t)(m0 + m) * ld + k0;
  FragU f;
#pragma unroll
  for (int v = 0; v < 8; ++v) {
    int kb = ((v & 4) << 2) + g * 8 + ((v & 3) << 1);
    f.u[v] = *reinterpret_cast<const uint32_t*>(row + kb);
  }
  return f.v;
}

__device__ __forceinline__ v16bf load_bfrag(const bf16_t* __restrict__ X, int ld,
                                            int n0, int k0) {
  int lane = threadIdx.x & 31;
  int n = lane & 15, kh = lane >> 4;
  const bf16_t* row = X + (size_t)(n0 + n) * ld + k0;
  FragU f;
#pragma unroll
  for (int v = 0; v < 8; ++v)
    f.u[v] = *reinterpret_cast<const uint32_t*>(row + ((v & 4) << 2) + kh * 8 + ((v & 3) << 1));
  return f.v;
}

__device__ __forceinline__ v8f wmma_bf16(v16bf a, v16bf b, v8f c) {
  return __builtin_amdgcn_wmma_f32_16x16x32_bf16(false, a, false, b, (short)0, c,
                                                 false, false);
}

__device__ __forceinline__ v8f zero8() {
  v8f z = {0.f, 0.f, 0.f, 0.f, 0.f, 0.f, 0.f, 0.f};
  return z;
}

// CDNA5 async copy: 16B global -> LDS, tracked by ASYNCcnt (CDNA5 ISA).
__device__ __forceinline__ void async_copy16(uint32_t lds_off, const void* gaddr) {
  asm volatile("global_load_async_to_lds_b128 %0, %1, off"
               :: "v"(lds_off), "v"(gaddr) : "memory");
}
__device__ __forceinline__ void wait_async0() {
  asm volatile("s_wait_asynccnt 0x0" ::: "memory");
}
__device__ __forceinline__ void wait_ds0() {
  asm volatile("s_wait_dscnt 0x0" ::: "memory");
}
__device__ __forceinline__ uint32_t lds_addr32(const void* p) {
  return (uint32_t)(uintptr_t)p;  // low 32 bits of flat addr == DS address
}

// ---------------------------------------------------------------------------
// Kernel 0: fp32 -> bf16 conversion
// ---------------------------------------------------------------------------
__global__ void cvt_f32_bf16(const float* __restrict__ in, bf16_t* __restrict__ out,
                             int n) {
  int i = blockIdx.x * blockDim.x + threadIdx.x;
  if (i < n) out[i] = (bf16_t)in[i];
}

// ---------------------------------------------------------------------------
// Kernel 1: all_bwd = x @ W_proj^T, split into
//   inp_t[b][h][w][k]  and  out_t[b][h][k][w] (transposed for contiguous
//   B-fragments in the scores@out GEMM). Wave computes a 16x64 tile.
// ---------------------------------------------------------------------------
__global__ void proj_gemm(const bf16_t* __restrict__ xb,
                          const bf16_t* __restrict__ wpb,
                          bf16_t* __restrict__ inp_t,
                          bf16_t* __restrict__ out_t) {
  const int C = 1024, Wn = 2048, H = 16, KH = 64;
  int wave = blockIdx.x * (blockDim.x >> 5) + (threadIdx.x >> 5);
  const int tiles_n = (2 * C) / 64;  // 32
  int tm = wave / tiles_n, tn = wave % tiles_n;
  int m0 = tm * 16, n0 = tn * 64;

  v8f acc[4] = {zero8(), zero8(), zero8(), zero8()};
  for (int k0 = 0; k0 < C; k0 += 32) {
    v16bf a = load_afrag(xb, C, m0, k0);
    if (k0 + 32 < C) {
      // near-cache (WGP-scope) prefetch of next k-tile: A row + B rows
      __builtin_prefetch(xb + (size_t)m0 * C + k0 + 32, 0, 3);
      __builtin_prefetch(wpb + (size_t)n0 * C + k0 + 32, 0, 3);
    }
#pragma unroll
    for (int j = 0; j < 4; ++j) {
      v16bf b = load_bfrag(wpb, C, n0 + j * 16, k0);
      acc[j] = wmma_bf16(a, b, acc[j]);
    }
  }

  int lane = threadIdx.x & 31;
  int n = lane & 15, mh = lane >> 4;
  int bidx = m0 / Wn;
  int wrow0 = m0 % Wn;
  int head = n0 / 64;  // 0..31; <16 -> inp half, >=16 -> out half
#pragma unroll
  for (int j = 0; j < 4; ++j) {
    int colh = j * 16 + n;
#pragma unroll
    for (int r = 0; r < 8; ++r) {
      int wrow = wrow0 + r + 8 * mh;
      bf16_t bv = (bf16_t)acc[j][r];
      if (head < H) {
        inp_t[(((size_t)bidx * H + head) * Wn + wrow) * KH + colh] = bv;
      } else {
        out_t[(((size_t)bidx * H + (head - H)) * KH + colh) * Wn + wrow] = bv;
      }
    }
  }
}

// ---------------------------------------------------------------------------
// Kernel 2: causal scores = (Q Q^T)/8 (tril, zeros, NO softmax),
//           nudged = scores @ out.  128-thread block = 4 waves = 64 q-rows of
//           one head; each 32-wide s-chunk is async-staged into LDS once and
//           shared by all 4 q-tiles (4x L2-traffic reduction).
// ---------------------------------------------------------------------------
__global__ void attn_kernel(const bf16_t* __restrict__ inp_t,
                            const bf16_t* __restrict__ out_t,
                            bf16_t* __restrict__ nudged) {
  const int Wn = 2048, H = 16, KH = 64, C = 1024;
  __shared__ __align__(16) bf16_t inpS[32 * 64];      // s-chunk of inp, [32][64]
  __shared__ __align__(16) bf16_t outS[64 * 32];      // s-chunk of out, [64][32]
  __shared__ __align__(16) bf16_t Sbuf[4 * 16 * 32];  // per-wave score tiles

  int bid = blockIdx.x;            // B*H*(W/64) = 1024
  int qb = bid & 31;               // 64-row q-block
  int h  = (bid >> 5) & 15;
  int b  = bid >> 9;
  int qb0 = qb * 64;
  const bf16_t* inp_h = inp_t + ((size_t)b * H + h) * Wn * KH;  // [W][64]
  const bf16_t* out_h = out_t + ((size_t)b * H + h) * KH * Wn;  // [64][W]

  int t = threadIdx.x;
  int qw = t >> 5;                 // wave id -> q-tile
  int q0 = qb0 + qw * 16;
  int lane = t & 31;
  int n = lane & 15, mh = lane >> 4;
  bf16_t* Sw = &Sbuf[qw * 16 * 32];

  uint32_t inpS_a = lds_addr32(&inpS[0]);
  uint32_t outS_a = lds_addr32(&outS[0]);

  v16bf a_lo = load_afrag(inp_h, KH, q0, 0);
  v16bf a_hi = load_afrag(inp_h, KH, q0, 32);
  v8f acc[4] = {zero8(), zero8(), zero8(), zero8()};
  const float scale = 0.125f;  // 1/sqrt(64)

  for (int s0 = 0; s0 < qb0 + 64; s0 += 32) {  // causal upper bound for block
    // ---- cooperative async stage: inp rows [s0,s0+32) (contiguous 4KB) ----
    const bf16_t* src = inp_h + (size_t)s0 * KH;
    for (int c = t; c < 256; c += 128)
      async_copy16(inpS_a + c * 16, src + c * 8);
    // ---- out_h rows k=0..63, cols [s0,s0+32): 64 rows x 64B -> outS[64][32]
    for (int c = t; c < 256; c += 128) {
      int row = c >> 2, col = (c & 3) * 8;
      async_copy16(outS_a + c * 16, out_h + (size_t)row * Wn + s0 + col);
    }
    wait_async0();
    __syncthreads();  // all waves' async data visible in LDS

    if (s0 < q0 + 16) {  // wave-uniform: skip fully-masked chunks
#pragma unroll
      for (int tt = 0; tt < 2; ++tt) {
        int scl = tt * 16;  // local s-tile offset
        v16bf b0 = load_bfrag(inpS, KH, scl, 0);
        v16bf b1 = load_bfrag(inpS, KH, scl, 32);
        v8f s_acc = zero8();
        s_acc = wmma_bf16(a_lo, b0, s_acc);
        s_acc = wmma_bf16(a_hi, b1, s_acc);
#pragma unroll
        for (int r = 0; r < 8; ++r) {
          int m = r + 8 * mh;
          int q = q0 + m, s = s0 + scl + n;
          float v = (s <= q) ? s_acc[r] * scale : 0.0f;
          Sw[m * 32 + scl + n] = (bf16_t)v;
        }
      }
      wait_ds0();  // wave-local ordering: S writes before S reads (LDS in-order)
      v16bf as = load_afrag(Sw, 32, 0, 0);
#pragma unroll
      for (int j = 0; j < 4; ++j) {
        v16bf bo = load_bfrag(outS, 32, j * 16, 0);
        acc[j] = wmma_bf16(as, bo, acc[j]);
      }
    }
    __syncthreads();  // protect LDS before next chunk's staging
  }

  // nudged as (B*W) x C row-major bf16
#pragma unroll
  for (int j = 0; j < 4; ++j)
#pragma unroll
    for (int r = 0; r < 8; ++r) {
      int m = r + 8 * mh;
      size_t row = (size_t)b * Wn + q0 + m;
      nudged[row * C + h * KH + j * 16 + n] = (bf16_t)acc[j][r];
    }
}

// ---------------------------------------------------------------------------
// Kernel 3: out = nudged @ W_out^T  (4096x1024)x(1024x1024), f32 output
// ---------------------------------------------------------------------------
__global__ void out_gemm(const bf16_t* __restrict__ nudged,
                         const bf16_t* __restrict__ wob,
                         float* __restrict__ outp) {
  const int C = 1024;
  int wave = blockIdx.x * (blockDim.x >> 5) + (threadIdx.x >> 5);
  const int tiles_n = C / 64;  // 16
  int tm = wave / tiles_n, tn = wave % tiles_n;
  int m0 = tm * 16, n0 = tn * 64;

  v8f acc[4] = {zero8(), zero8(), zero8(), zero8()};
  for (int k0 = 0; k0 < C; k0 += 32) {
    v16bf a = load_afrag(nudged, C, m0, k0);
    if (k0 + 32 < C) {
      __builtin_prefetch(nudged + (size_t)m0 * C + k0 + 32, 0, 3);
      __builtin_prefetch(wob + (size_t)n0 * C + k0 + 32, 0, 3);
    }
#pragma unroll
    for (int j = 0; j < 4; ++j) {
      v16bf b = load_bfrag(wob, C, n0 + j * 16, k0);
      acc[j] = wmma_bf16(a, b, acc[j]);
    }
  }
  int lane = threadIdx.x & 31;
  int n = lane & 15, mh = lane >> 4;
#pragma unroll
  for (int j = 0; j < 4; ++j)
#pragma unroll
    for (int r = 0; r < 8; ++r) {
      int m = r + 8 * mh;
      outp[(size_t)(m0 + m) * C + n0 + j * 16 + n] = acc[j][r];
    }
}

// ---------------------------------------------------------------------------
extern "C" void kernel_launch(void* const* d_in, const int* in_sizes, int n_in,
                              void* d_out, int out_size, void* d_ws, size_t ws_size,
                              hipStream_t stream) {
  const float* x      = (const float*)d_in[0];  // 2*2048*1024
  const float* W_proj = (const float*)d_in[1];  // 2048*1024
  const float* W_out  = (const float*)d_in[2];  // 1024*1024
  float* outp = (float*)d_out;

  const size_t M = 4096, C = 1024, N2 = 2048;
  char* ws = (char*)d_ws;
  bf16_t* xb   = (bf16_t*)ws; ws += M * C * sizeof(bf16_t);   // 8 MiB
  bf16_t* wpb  = (bf16_t*)ws; ws += N2 * C * sizeof(bf16_t);  // 4 MiB
  bf16_t* wob  = (bf16_t*)ws; ws += C * C * sizeof(bf16_t);   // 2 MiB
  bf16_t* inpt = (bf16_t*)ws; ws += M * C * sizeof(bf16_t);   // 8 MiB
  bf16_t* outt = (bf16_t*)ws; ws += M * C * sizeof(bf16_t);   // 8 MiB
  bf16_t* nud  = (bf16_t*)ws; ws += M * C * sizeof(bf16_t);   // 8 MiB

  cvt_f32_bf16<<<(int)((M * C + 255) / 256), 256, 0, stream>>>(x, xb, (int)(M * C));
  cvt_f32_bf16<<<(int)((N2 * C + 255) / 256), 256, 0, stream>>>(W_proj, wpb, (int)(N2 * C));
  cvt_f32_bf16<<<(int)((C * C + 255) / 256), 256, 0, stream>>>(W_out, wob, (int)(C * C));

  proj_gemm<<<2048, 128, 0, stream>>>(xb, wpb, inpt, outt);
  attn_kernel<<<1024, 128, 0, stream>>>(inpt, outt, nud);
  out_gemm<<<1024, 128, 0, stream>>>(nud, wob, outp);
}
